// PureQNet_87960930222350
// MI455X (gfx1250) — compile-verified, matrix operand order checked by reading, not run on hardware
//
#include <hip/hip_runtime.h>
#include <hip/hip_bf16.h>
#include <stdint.h>

typedef __attribute__((ext_vector_type(16))) _Float16 v16h;
typedef __attribute__((ext_vector_type(8)))  float    v8f;
typedef __attribute__((ext_vector_type(4)))  float    f4;

// ===========================================================================
// Kernel 1 (tiny, runs once per launch): build constant operands in d_ws.
//  - Simulate the 6-layer RX/CNOT-ring circuit on all 16 basis states to get
//    the 16x16 complex unitary U = Ur + i*Ui  (wire 0 = MSB of basis index).
//  - Fold sign-matrix S (16x10), head_w (10x10) and logit_scale into one
//    16->10 matrix A_head[o][k] = ls * sum_c S[k][c]*head_w[o][c].
//  - Emit Ur, Ui, A_head as WMMA f16 A-operand VGPR images (ISA 7.12.2
//    16-bit A 16x32 layout, K=16..31 zero-padded), 256 dwords each:
//      tiles[t*256 + lane*8 + v] = pack_f16(Mat[M][K0], Mat[M][K0+1])
//      M = lane%16, hi = lane/16, K0 = (v<4 ? 2v : 16+2(v-4)) + 8*hi
//  - biasls[16] = ls*head_b (zero padded).
// ===========================================================================
__global__ __launch_bounds__(64) void qnet_pre(
    const float* __restrict__ wts,   // [6,4]
    const float* __restrict__ hw,    // [10,10]
    const float* __restrict__ hb,    // [10]
    const float* __restrict__ lsp,   // [1]
    uint32_t* __restrict__ tiles,    // 768 dwords
    float* __restrict__ biasls)      // 16 floats
{
  __shared__ float Ur[16][16];
  __shared__ float Ui[16][16];
  const int tid = threadIdx.x;

  if (tid < 16) {
    float sr[16], si[16], nr[16], ni[16];
    for (int i = 0; i < 16; ++i) { sr[i] = 0.f; si[i] = 0.f; }
    sr[tid] = 1.f;                          // column 'tid' of U
    const int rc[4] = {0, 1, 2, 3};
    const int rt[4] = {1, 2, 3, 0};
    for (int l = 0; l < 6; ++l) {
      for (int q = 0; q < 4; ++q) {         // RX(theta) on wire q
        float th = 0.5f * wts[l * 4 + q];
        float c = cosf(th), s = sinf(th);
        int m = 1 << (3 - q);               // wire q <-> bit (3-q)
        for (int i = 0; i < 16; ++i) {
          nr[i] = c * sr[i] + s * si[i ^ m];   // (-i sin) * (a+bi) = s*b - i*s*a
          ni[i] = c * si[i] - s * sr[i ^ m];
        }
        for (int i = 0; i < 16; ++i) { sr[i] = nr[i]; si[i] = ni[i]; }
      }
      for (int e = 0; e < 4; ++e) {         // CNOT ring
        int mc = 1 << (3 - rc[e]);
        int mt = 1 << (3 - rt[e]);
        for (int i = 0; i < 16; ++i) {
          int srcI = (i & mc) ? (i ^ mt) : i;
          nr[i] = sr[srcI]; ni[i] = si[srcI];
        }
        for (int i = 0; i < 16; ++i) { sr[i] = nr[i]; si[i] = ni[i]; }
      }
    }
    for (int i = 0; i < 16; ++i) { Ur[i][tid] = sr[i]; Ui[i][tid] = si[i]; }
  }
  __syncthreads();

  const float ls = lsp[0];
  for (int e = tid; e < 768; e += 64) {
    int t = e >> 8, r = e & 255, lane = r >> 3, v = r & 7;
    int M = lane & 15, hi = lane >> 4;
    int K0 = ((v < 4) ? (2 * v) : (16 + 2 * (v - 4))) + 8 * hi;
    float v0 = 0.f, v1 = 0.f;
    if (K0 < 16) {
      if (t == 0)      { v0 = Ur[M][K0]; v1 = Ur[M][K0 + 1]; }
      else if (t == 1) { v0 = Ui[M][K0]; v1 = Ui[M][K0 + 1]; }
      else if (M < 10) {
        #pragma unroll
        for (int half = 0; half < 2; ++half) {
          int k = K0 + half;
          float z0 = 1.f - 2.f * ((k >> 3) & 1);
          float z1 = 1.f - 2.f * ((k >> 2) & 1);
          float z2 = 1.f - 2.f * ((k >> 1) & 1);
          float z3 = 1.f - 2.f * (k & 1);
          float sc[10] = {z0, z1, z2, z3, z0*z1, z0*z2, z0*z3, z1*z2, z1*z3, z2*z3};
          float acc = 0.f;
          for (int c2 = 0; c2 < 10; ++c2) acc += sc[c2] * hw[M * 10 + c2];
          if (half == 0) v0 = ls * acc; else v1 = ls * acc;
        }
      }
    }
    union { _Float16 h[2]; uint32_t u; } cv;
    cv.h[0] = (_Float16)v0; cv.h[1] = (_Float16)v1;
    tiles[e] = cv.u;
  }
  if (tid < 16) biasls[tid] = (tid < 10) ? ls * hb[tid] : 0.f;
}

// ===========================================================================
// Kernel 2: streaming main kernel. 256 threads = 8 waves; each wave owns 16
// samples (128 samples / block). HBM-bound: 128 MB in, 1.3 MB out.
// ===========================================================================
__global__ __launch_bounds__(256) void qnet_main(
    const float* __restrict__ x,          // [B,1,32,32]
    const uint32_t* __restrict__ tiles,   // A-operand images (Ur,Ui,head)
    const float* __restrict__ biasls,     // 16
    float* __restrict__ out,              // [B,10]
    int nsamp)
{
  __shared__ float amp[8][256];           // per-wave 16(basis) x 16(sample)
  const int tid  = threadIdx.x;
  const int wv   = tid >> 5;
  const int lane = tid & 31;
  const int base_s = (blockIdx.x * 8 + wv) * 16;
  const bool active = (base_s < nsamp);
  const int N = lane & 15, hi = lane >> 4;

  // ---- phase 1: pool 32x32 -> 4x4 block sums, L2-normalize, stash in LDS.
  // Lane l, iter it: float4 at linear idx 128*it + 4*l (coalesced 512B/wave).
  // That float4 lies fully inside pool cell (r = it/2, c = (l%8)/2); the 1/64
  // mean factor cancels under L2 normalization so plain sums suffice.
  if (active) {
    for (int s = 0; s < 16; ++s) {
      const f4* px = (const f4*)(x + (size_t)(base_s + s) * 1024);
      float acc[4] = {0.f, 0.f, 0.f, 0.f};
      #pragma unroll
      for (int it = 0; it < 8; ++it) {
        f4 vv = __builtin_nontemporal_load(px + it * 32 + lane);
        acc[it >> 1] += (vv.x + vv.y) + (vv.z + vv.w);
      }
      // reduce over lane bits 0,3,4 (lanes sharing pool column group c)
      #pragma unroll
      for (int r = 0; r < 4; ++r) {
        acc[r] += __shfl_xor(acc[r], 1, 32);
        acc[r] += __shfl_xor(acc[r], 8, 32);
        acc[r] += __shfl_xor(acc[r], 16, 32);
      }
      // squared norm: sum over r locally, over c via lane bits 1,2
      float nrm = acc[0]*acc[0] + acc[1]*acc[1] + acc[2]*acc[2] + acc[3]*acc[3];
      nrm += __shfl_xor(nrm, 2, 32);
      nrm += __shfl_xor(nrm, 4, 32);
      float inv = rsqrtf(nrm);
      if ((lane & 1) == 0 && lane < 8) {          // one writer per c-group
        int c = lane >> 1;
        #pragma unroll
        for (int r = 0; r < 4; ++r)
          amp[wv][(4 * r + c) * 16 + s] = acc[r] * inv;
      }
    }
  }
  __syncthreads();            // all threads reach this (keeps barrier safe)
  if (!active) return;

  // ---- phase 2: WMMA. A operands preloaded as exact VGPR images.
  v16h a_ur, a_ui, a_hd;
  {
    uint32_t* pu = reinterpret_cast<uint32_t*>(&a_ur);
    uint32_t* pv = reinterpret_cast<uint32_t*>(&a_ui);
    uint32_t* ph = reinterpret_cast<uint32_t*>(&a_hd);
    #pragma unroll
    for (int v = 0; v < 8; ++v) {
      pu[v] = tiles[lane * 8 + v];
      pv[v] = tiles[256 + lane * 8 + v];
      ph[v] = tiles[512 + lane * 8 + v];
    }
  }

  // B operand: amplitudes, lane = sample column N, K striped like A layout,
  // K=16..31 zero (unitary is only 16x16).
  v16h b;
  #pragma unroll
  for (int v = 0; v < 8; ++v) {
    if (v < 4) {
      int K0 = 2 * v + 8 * hi;
      b[2 * v]     = (_Float16)amp[wv][K0 * 16 + N];
      b[2 * v + 1] = (_Float16)amp[wv][(K0 + 1) * 16 + N];
    } else {
      b[2 * v] = (_Float16)0.f; b[2 * v + 1] = (_Float16)0.f;
    }
  }

  v8f zc = {0.f, 0.f, 0.f, 0.f, 0.f, 0.f, 0.f, 0.f};
  // Re/Im halves of the state for all 16 samples at once.
  v8f dre = __builtin_amdgcn_wmma_f32_16x16x32_f16(false, a_ur, false, b,
                                                   (short)0, zc, false, false);
  v8f dim = __builtin_amdgcn_wmma_f32_16x16x32_f16(false, a_ui, false, b,
                                                   (short)0, zc, false, false);

  // probs: D layout (M = u + 8*hi, N = lane%16) coincides exactly with the
  // B layout needed for the head GEMM -> no cross-lane movement.
  v16h b2;
  #pragma unroll
  for (int u = 0; u < 8; ++u)
    b2[u] = (_Float16)(dre[u] * dre[u] + dim[u] * dim[u]);
  #pragma unroll
  for (int u = 8; u < 16; ++u) b2[u] = (_Float16)0.f;

  v8f cb;                                   // ls*head_b via C operand
  #pragma unroll
  for (int v = 0; v < 8; ++v) cb[v] = biasls[v + 8 * hi];
  v8f dout = __builtin_amdgcn_wmma_f32_16x16x32_f16(false, a_hd, false, b2,
                                                    (short)0, cb, false, false);

  // store: lane holds sample N, VGPR v holds output row o = v + 8*hi (o<10)
  const int sg = base_s + N;
  #pragma unroll
  for (int v = 0; v < 8; ++v) {
    int o = v + 8 * hi;
    if (o < 10)
      __builtin_nontemporal_store(dout[v], out + (size_t)sg * 10 + o);
  }
}

// ===========================================================================
extern "C" void kernel_launch(void* const* d_in, const int* in_sizes, int n_in,
                              void* d_out, int out_size, void* d_ws, size_t ws_size,
                              hipStream_t stream) {
  const float* x   = (const float*)d_in[0];   // [32768,1,32,32]
  const float* wts = (const float*)d_in[1];   // [6,4]
  const float* hw  = (const float*)d_in[2];   // [10,10]
  const float* hb  = (const float*)d_in[3];   // [10]
  const float* ls  = (const float*)d_in[4];   // [1]
  float* out = (float*)d_out;

  uint32_t* tiles  = (uint32_t*)d_ws;         // 768 dwords
  float*    biasls = ((float*)d_ws) + 768;    // 16 floats

  const int B = in_sizes[0] / 1024;

  qnet_pre<<<1, 64, 0, stream>>>(wts, hw, hb, ls, tiles, biasls);

  const int nblocks = (B + 127) / 128;        // 8 waves x 16 samples / block
  qnet_main<<<nblocks, 256, 0, stream>>>(x, tiles, biasls, out, B);
}
